// Attention_1185410973857
// MI455X (gfx1250) — compile-verified, matrix operand order checked by reading, not run on hardware
//
#include <hip/hip_runtime.h>
#include <hip/hip_bf16.h>
#include <cstddef>

typedef __attribute__((ext_vector_type(16))) __bf16 v16bf;
typedef __attribute__((ext_vector_type(8)))  __bf16 v8bf;
typedef __attribute__((ext_vector_type(8)))  float  v8f;
typedef __attribute__((ext_vector_type(4)))  float  v4f;

#define WMMA_BF16(a, b, c) \
  __builtin_amdgcn_wmma_f32_16x16x32_bf16(false, (a), false, (b), (short)0, (c), false, false)

// Assemble a 16-half WMMA operand from two contiguous 16-byte LDS chunks.
__device__ __forceinline__ v16bf frag16(const __bf16* lo, const __bf16* hi) {
  v8bf a = *(const v8bf*)lo;
  v8bf b = *(const v8bf*)hi;
  return __builtin_shufflevector(a, b, 0, 1, 2, 3, 4, 5, 6, 7,
                                 8, 9, 10, 11, 12, 13, 14, 15);
}

__device__ __forceinline__ v8bf pack8(v4f f0, v4f f1) {
  v8bf o;
#pragma unroll
  for (int i = 0; i < 4; ++i) {
    o[i]     = (__bf16)f0[i];
    o[4 + i] = (__bf16)f1[i];
  }
  return o;
}

// ---------------------------------------------------------------------------
// im2col for 2x2 stride-2 conv:  A[b*OH*OW + oy*OW + ox, p*256 + ic]
// ---------------------------------------------------------------------------
__global__ void im2col_kernel(const float* __restrict__ t, float* __restrict__ A,
                              int B, int H, int W) {
  int OH = H >> 1, OW = W >> 1;
  size_t total = (size_t)B * OH * OW * 1024;
  size_t idx = (size_t)blockIdx.x * blockDim.x + threadIdx.x;
  if (idx >= total) return;
  int k  = (int)(idx & 1023);
  size_t m = idx >> 10;
  int p  = k >> 8;
  int ic = k & 255;
  int ky = p >> 1, kx = p & 1;
  int hw = OH * OW;
  int b  = (int)(m / hw);
  int rem = (int)(m % hw);
  int oy = rem / OW, ox = rem % OW;
  int iy = 2 * oy + ky, ix = 2 * ox + kx;
  A[idx] = t[(((size_t)b * H + iy) * W + ix) * 256 + ic];
}

// sr_w [O=256][I=256][2][2] -> W'[p*256+ic][c]
__global__ void wT_kernel(const float* __restrict__ w, float* __restrict__ wt) {
  int idx = blockIdx.x * blockDim.x + threadIdx.x;
  if (idx >= 1024 * 256) return;
  int c  = idx & 255;
  int k  = idx >> 8;       // k = p*256 + ic
  int p  = k >> 8;
  int ic = k & 255;
  wt[idx] = w[((size_t)c * 256 + ic) * 4 + p];
}

// ---------------------------------------------------------------------------
// Generic bf16 WMMA GEMM: C[M,N] = A[M,K] @ B[K,N] (+ bias[N])
// 128 threads = 4 waves; tile 64x64, K-chunk 32. B tile stored transposed in
// LDS so every WMMA operand loads as two ds_load_b128.
// ---------------------------------------------------------------------------
__global__ __launch_bounds__(128)
void gemm_bf16_kernel(const float* __restrict__ A, const float* __restrict__ B,
                      const float* __restrict__ bias, float* __restrict__ C,
                      int M, int N, int K) {
  __shared__ __align__(16) __bf16 At[64][32];    // [row][k]
  __shared__ __align__(16) __bf16 BtT[64][32];   // [col][k]  (transposed tile)
  const int tid  = threadIdx.x;
  const int wave = tid >> 5;
  const int lane = tid & 31;
  const int lr   = lane & 15;
  const int hi   = lane >> 4;
  const int kb   = hi * 8;
  const int m0   = blockIdx.x * 64;
  const int n0   = blockIdx.y * 64;

  v8f acc[4] = {};

  for (int k0 = 0; k0 < K; k0 += 32) {
    // A tile: 64x32 = 256 8-elem chunks; 2 per thread; b128 in, b128 out
#pragma unroll
    for (int cc = 0; cc < 2; ++cc) {
      int cid = tid + cc * 128;
      int r = cid >> 2, c8 = (cid & 3) * 8;
      const float* src = &A[(size_t)(m0 + r) * K + k0 + c8];
      v4f f0 = *(const v4f*)src;
      v4f f1 = *(const v4f*)(src + 4);
      *(v8bf*)&At[r][c8] = pack8(f0, f1);
    }
    // B tile (transposed store): per chunk, 8 coalesced dword loads along K
#pragma unroll
    for (int cc = 0; cc < 2; ++cc) {
      int cid = tid + cc * 128;
      int c = cid >> 2, r8 = (cid & 3) * 8;
      v8bf o;
#pragma unroll
      for (int i = 0; i < 8; ++i)
        o[i] = (__bf16)B[(size_t)(k0 + r8 + i) * N + n0 + c];
      *(v8bf*)&BtT[c][r8] = o;
    }
    if (k0 + 32 < K) {
      __builtin_prefetch(&A[(size_t)(m0 + (tid >> 2)) * K + k0 + 32], 0, 0);
      __builtin_prefetch(&B[(size_t)(k0 + 32 + (tid >> 2)) * N + n0], 0, 0);
    }
    __syncthreads();

    const int ar = wave * 16 + lr;
    v16bf a = frag16(&At[ar][kb], &At[ar][16 + kb]);
#pragma unroll
    for (int t = 0; t < 4; ++t) {
      int c = t * 16 + lr;
      v16bf b = frag16(&BtT[c][kb], &BtT[c][16 + kb]);
      acc[t] = WMMA_BF16(a, b, acc[t]);
    }
    __syncthreads();
  }

#pragma unroll
  for (int t = 0; t < 4; ++t) {
    int col = n0 + t * 16 + lr;
    float bv = bias ? bias[col] : 0.0f;
#pragma unroll
    for (int j = 0; j < 8; ++j) {
      int row = m0 + wave * 16 + hi * 8 + j;
      C[(size_t)row * N + col] = acc[t][j] + bv;
    }
  }
}

// ---------------------------------------------------------------------------
// Row LayerNorm over 256 channels (in place)
// ---------------------------------------------------------------------------
__global__ __launch_bounds__(256)
void ln_kernel(float* __restrict__ red, const float* __restrict__ g,
               const float* __restrict__ b, int M) {
  __shared__ float sbuf[256];
  int row = blockIdx.x;
  int c = threadIdx.x;
  float v = red[(size_t)row * 256 + c];
  sbuf[c] = v;
  __syncthreads();
  for (int s = 128; s > 0; s >>= 1) { if (c < s) sbuf[c] += sbuf[c + s]; __syncthreads(); }
  float mean = sbuf[0] * (1.0f / 256.0f);
  __syncthreads();
  float d = v - mean;
  sbuf[c] = d * d;
  __syncthreads();
  for (int s = 128; s > 0; s >>= 1) { if (c < s) sbuf[c] += sbuf[c + s]; __syncthreads(); }
  float var = sbuf[0] * (1.0f / 256.0f);
  red[(size_t)row * 256 + c] = d * rsqrtf(var + 1e-5f) * g[c] + b[c];
}

// mean over 4 support batches of KV rows
__global__ void meankv_kernel(const float* __restrict__ kvy, float* __restrict__ out) {
  int idx = blockIdx.x * blockDim.x + threadIdx.x;
  if (idx >= 256 * 512) return;
  int n = idx >> 9;
  int j = idx & 511;
  float s = 0.f;
#pragma unroll
  for (int b = 0; b < 4; ++b) s += kvy[((size_t)(b * 256 + n)) * 512 + j];
  out[idx] = 0.25f * s;
}

// ---------------------------------------------------------------------------
// Flash attention: grid = (nq/64, heads). Wave handles 16 q-rows.
// KV matrices [rows,512]: K at cols head*32+d, V at 256+head*32+d.
// Keys = concat(KV1[0:len1], KV2[0:len2]); both multiples of 32.
// V staged transposed so its column fragments load as ds_load_b128.
// ---------------------------------------------------------------------------
__global__ __launch_bounds__(128)
void attn_kernel(const float* __restrict__ Q,
                 const float* __restrict__ KV1, int len1,
                 const float* __restrict__ KV2, int len2,
                 float* __restrict__ O) {
  __shared__ __align__(16) __bf16 Kt[32][32];        // [key][d]
  __shared__ __align__(16) __bf16 VtT[32][32];       // [d][key] (transposed)
  __shared__ __align__(16) __bf16 Pt[4][16][32];     // per-wave P scratch

  const int tid  = threadIdx.x;
  const int wave = tid >> 5;
  const int lane = tid & 31;
  const int lr   = lane & 15;
  const int hi   = lane >> 4;
  const int kb   = hi * 8;
  const int head = blockIdx.y;
  const int q0   = blockIdx.x * 64 + wave * 16;
  const float scale = 0.17677669529663687f;  // 1/sqrt(32)

  // Q fragment, pre-scaled (b128 global loads)
  v16bf qa;
  {
    const float* qp = &Q[(size_t)(q0 + lr) * 256 + head * 32];
    v4f f0 = *(const v4f*)(qp + kb);
    v4f f1 = *(const v4f*)(qp + kb + 4);
    v4f f2 = *(const v4f*)(qp + 16 + kb);
    v4f f3 = *(const v4f*)(qp + 16 + kb + 4);
#pragma unroll
    for (int i = 0; i < 4; ++i) {
      qa[i]      = (__bf16)(f0[i] * scale);
      qa[4 + i]  = (__bf16)(f1[i] * scale);
      qa[8 + i]  = (__bf16)(f2[i] * scale);
      qa[12 + i] = (__bf16)(f3[i] * scale);
    }
  }

  float mrow[8], lrow[8];
#pragma unroll
  for (int j = 0; j < 8; ++j) { mrow[j] = -1e30f; lrow[j] = 0.f; }
  v8f acc0 = {}, acc1 = {};

  const int ntiles = (len1 + len2) >> 5;
  for (int kt = 0; kt < ntiles; ++kt) {
    int kvbase = kt * 32;
    // K tile: row-major; one 8-elem chunk per thread (b128 in/out)
    {
      int r = tid >> 2, c8 = (tid & 3) * 8;
      int key = kvbase + r;
      const float* src; int kr;
      if (key < len1) { src = KV1; kr = key; } else { src = KV2; kr = key - len1; }
      const float* p = &src[(size_t)kr * 512 + head * 32 + c8];
      v4f f0 = *(const v4f*)p;
      v4f f1 = *(const v4f*)(p + 4);
      *(v8bf*)&Kt[r][c8] = pack8(f0, f1);
    }
    // V tile transposed: VtT[d][key]; gather 8 keys for one d, b128 out
    {
      int d = tid >> 2, k8 = (tid & 3) * 8;
      v8bf o;
#pragma unroll
      for (int i = 0; i < 8; ++i) {
        int key = kvbase + k8 + i;
        const float* src; int kr;
        if (key < len1) { src = KV1; kr = key; } else { src = KV2; kr = key - len1; }
        o[i] = (__bf16)src[(size_t)kr * 512 + head * 32 + 256 + d];
      }
      *(v8bf*)&VtT[d][k8] = o;
    }
    __syncthreads();

    // scores: S0 = Q @ K[0:16]^T, S1 = Q @ K[16:32]^T
    v16bf b0 = frag16(&Kt[lr][kb],      &Kt[lr][16 + kb]);
    v16bf b1 = frag16(&Kt[16 + lr][kb], &Kt[16 + lr][16 + kb]);
    v8f s0 = {}, s1 = {};
    s0 = WMMA_BF16(qa, b0, s0);
    s1 = WMMA_BF16(qa, b1, s1);

    // online softmax; row m lives in c[m&7] across the 16 lanes of one half
#pragma unroll
    for (int j = 0; j < 8; ++j) {
      float v0 = s0[j], v1 = s1[j];
      float mx = fmaxf(v0, v1);
#pragma unroll
      for (int m = 1; m < 16; m <<= 1) mx = fmaxf(mx, __shfl_xor(mx, m, 32));
      float mnew  = fmaxf(mrow[j], mx);
      float alpha = __expf(mrow[j] - mnew);
      float p0 = __expf(v0 - mnew), p1 = __expf(v1 - mnew);
      float ps = p0 + p1;
#pragma unroll
      for (int m = 1; m < 16; m <<= 1) ps += __shfl_xor(ps, m, 32);
      lrow[j] = lrow[j] * alpha + ps;
      mrow[j] = mnew;
      acc0[j] *= alpha;
      acc1[j] *= alpha;
      int prow = hi * 8 + j;
      Pt[wave][prow][lr]      = (__bf16)p0;
      Pt[wave][prow][16 + lr] = (__bf16)p1;
    }
    __syncthreads();

    // O += P @ V  (two 16-col halves of d)
    v16bf pa  = frag16(&Pt[wave][lr][kb], &Pt[wave][lr][16 + kb]);
    v16bf vb0 = frag16(&VtT[lr][kb],      &VtT[lr][16 + kb]);
    v16bf vb1 = frag16(&VtT[16 + lr][kb], &VtT[16 + lr][16 + kb]);
    acc0 = WMMA_BF16(pa, vb0, acc0);
    acc1 = WMMA_BF16(pa, vb1, acc1);
    __syncthreads();
  }

#pragma unroll
  for (int j = 0; j < 8; ++j) {
    float inv = 1.0f / lrow[j];
    int row = q0 + hi * 8 + j;
    size_t base = (size_t)row * 256 + head * 32;
    O[base + lr]      = acc0[j] * inv;
    O[base + 16 + lr] = acc1[j] * inv;
  }
}

// ---------------------------------------------------------------------------
extern "C" void kernel_launch(void* const* d_in, const int* in_sizes, int n_in,
                              void* d_out, int out_size, void* d_ws, size_t ws_size,
                              hipStream_t stream) {
  const float* x      = (const float*)d_in[0];   // [1,6400,256]
  const float* y      = (const float*)d_in[1];   // [4,1024,256]
  const float* Wq     = (const float*)d_in[2];   // [256,256]
  const float* Wkv    = (const float*)d_in[3];   // [256,512]
  const float* sr_w   = (const float*)d_in[4];   // [256,256,2,2]
  const float* sr_b   = (const float*)d_in[5];   // [256]
  const float* ln_g   = (const float*)d_in[6];
  const float* ln_b   = (const float*)d_in[7];
  const float* proj_w = (const float*)d_in[8];   // [256,256]
  const float* proj_b = (const float*)d_in[9];
  float* out = (float*)d_out;

  float* ws = (float*)d_ws;
  float* Qx   = ws;               // 6400*256
  float* Qy   = Qx   + 1638400;   // 4096*256
  float* Ax   = Qy   + 1048576;   // 1600*1024 im2col(x)
  float* Ay   = Ax   + 1638400;   // 1024*1024 im2col(y)
  float* WsrT = Ay   + 1048576;   // 1024*256
  float* RedX = WsrT + 262144;    // 1600*256
  float* RedY = RedX + 409600;    // 1024*256
  float* KVx  = RedY + 262144;    // 1600*512
  float* KVy  = KVx  + 819200;    // 1024*512
  float* KVym = KVy  + 524288;    // 256*512
  float* Ox   = KVym + 131072;    // 6400*256
  float* Oy   = Ox   + 1638400;   // 4096*256

  im2col_kernel<<<(1600 * 1024 + 255) / 256, 256, 0, stream>>>(x, Ax, 1, 80, 80);
  im2col_kernel<<<(1024 * 1024 + 255) / 256, 256, 0, stream>>>(y, Ay, 4, 32, 32);
  wT_kernel<<<(1024 * 256 + 255) / 256, 256, 0, stream>>>(sr_w, WsrT);

  gemm_bf16_kernel<<<dim3(100, 4), 128, 0, stream>>>(x, Wq, nullptr, Qx, 6400, 256, 256);
  gemm_bf16_kernel<<<dim3(64, 4), 128, 0, stream>>>(y, Wq, nullptr, Qy, 4096, 256, 256);

  gemm_bf16_kernel<<<dim3(25, 4), 128, 0, stream>>>(Ax, WsrT, sr_b, RedX, 1600, 256, 1024);
  gemm_bf16_kernel<<<dim3(16, 4), 128, 0, stream>>>(Ay, WsrT, sr_b, RedY, 1024, 256, 1024);
  ln_kernel<<<1600, 256, 0, stream>>>(RedX, ln_g, ln_b, 1600);
  ln_kernel<<<1024, 256, 0, stream>>>(RedY, ln_g, ln_b, 1024);

  gemm_bf16_kernel<<<dim3(25, 8), 128, 0, stream>>>(RedX, Wkv, nullptr, KVx, 1600, 512, 256);
  gemm_bf16_kernel<<<dim3(16, 8), 128, 0, stream>>>(RedY, Wkv, nullptr, KVy, 1024, 512, 256);
  meankv_kernel<<<512, 256, 0, stream>>>(KVy, KVym);

  attn_kernel<<<dim3(100, 8), 128, 0, stream>>>(Qx, KVx, 1600, KVym, 256, Ox);
  for (int b = 0; b < 4; ++b) {
    attn_kernel<<<dim3(16, 8), 128, 0, stream>>>(Qy + (size_t)b * 262144,
                                                 KVx, 1600,
                                                 KVy + (size_t)b * 131072, 256,
                                                 Oy + (size_t)b * 262144);
  }

  gemm_bf16_kernel<<<dim3(100, 4), 128, 0, stream>>>(Ox, proj_w, proj_b, out, 6400, 256, 256);
  gemm_bf16_kernel<<<dim3(64, 4), 128, 0, stream>>>(Oy, proj_w, proj_b, out + 1638400, 4096, 256, 256);
}
